// TransformerBlock_14285061226822
// MI455X (gfx1250) — matrix operand order repaired and checked by reading, not compiled, not run on hardware
//
#include <hip/hip_runtime.h>
#include <hip/hip_bf16.h>

typedef __bf16 bf16;
typedef __attribute__((ext_vector_type(8)))  bf16  v8bf;
typedef __attribute__((ext_vector_type(16))) bf16  v16bf;
typedef __attribute__((ext_vector_type(8)))  float v8f;

#define D_EMB 1024
#define NH    16
#define DH    64
#define TSEQ  2048
#define NB    2
#define BT    (NB*TSEQ)      // 4096
#define DFF   4096
#define EPS   1e-5f

// ---------------- WMMA helpers ----------------

__device__ __forceinline__ v8f vzero8() {
  v8f v;
#pragma unroll
  for (int i = 0; i < 8; ++i) v[i] = 0.0f;
  return v;
}

__device__ __forceinline__ v8f wmma_bf16(v16bf a, v16bf b, v8f c) {
  return __builtin_amdgcn_wmma_f32_16x16x32_bf16(
      /*neg_a=*/false, a, /*neg_b=*/false, b,
      /*c_mod=*/(short)0, c, /*reuse_a=*/false, /*reuse_b=*/false);
}

// A-matrix (16x32, row-major source). Lane holds row M=lane&15.
// Element e -> K = (hi<<4) + e: one contiguous 32B run (two 16B loads).
__device__ __forceinline__ v16bf load_a16(const bf16* __restrict__ row, int hi) {
  v8bf lo8 = *(const v8bf*)(row + hi * 16);
  v8bf hi8 = *(const v8bf*)(row + hi * 16 + 8);
  v16bf a;
#pragma unroll
  for (int i = 0; i < 8; ++i) { a[i] = lo8[i]; a[i + 8] = hi8[i]; }
  return a;
}

// One k-chunk of operands for a 32x64x32 MAC step (ping-pong set).
struct OpSet {
  v16bf a0, a1;
  v16bf b[4];
};

// 16-lane (column-dimension) reductions within each half-wave.
__device__ __forceinline__ float redmax16(float v) {
#pragma unroll
  for (int m = 1; m < 16; m <<= 1) v = fmaxf(v, __shfl_xor(v, m, 32));
  return v;
}
__device__ __forceinline__ float redsum16(float v) {
#pragma unroll
  for (int m = 1; m < 16; m <<= 1) v += __shfl_xor(v, m, 32);
  return v;
}

// ---------------- weight prep (fp32 -> bf16, transposed) ----------------

// Wq/Wk/Wv: [H, D, DH] -> [H, DH, D]
__global__ __launch_bounds__(256) void cvt_qkv_t(const float* __restrict__ W,
                                                 bf16* __restrict__ Wt) {
  int idx = blockIdx.x * 256 + threadIdx.x;          // H*D*DH = 1,048,576
  if (idx >= NH * D_EMB * DH) return;
  int h = idx >> 16;
  int rem = idx & 65535;
  int d = rem >> 6;
  int e = rem & 63;
  Wt[(((size_t)h * DH + e) << 10) + d] = (bf16)W[idx];
}

// W1: [D, DFF] -> [DFF, D]
__global__ __launch_bounds__(256) void cvt_w1_t(const float* __restrict__ W,
                                                bf16* __restrict__ Wt) {
  int idx = blockIdx.x * 256 + threadIdx.x;          // 4,194,304
  if (idx >= D_EMB * DFF) return;
  int d = idx >> 12;
  int f = idx & 4095;
  Wt[((size_t)f << 10) + d] = (bf16)W[idx];
}

// W2: [DFF, D] -> [D, DFF]
__global__ __launch_bounds__(256) void cvt_w2_t(const float* __restrict__ W,
                                                bf16* __restrict__ Wt) {
  int idx = blockIdx.x * 256 + threadIdx.x;          // 4,194,304
  if (idx >= DFF * D_EMB) return;
  int f = idx >> 10;
  int d = idx & 1023;
  Wt[((size_t)d << 12) + f] = (bf16)W[idx];
}

// ---------------- LayerNorm 1 (x -> bf16 h) ----------------

__global__ __launch_bounds__(256) void ln1_fwd(const float* __restrict__ x,
                                               const float* __restrict__ gamma,
                                               const float* __restrict__ beta,
                                               bf16* __restrict__ out) {
  __shared__ float rs[256], rs2[256];
  int row = blockIdx.x;
  int tid = threadIdx.x;
  const float* xr = x + ((size_t)row << 10);
  float s = 0.f, s2 = 0.f;
  for (int i = tid; i < D_EMB; i += 256) { float v = xr[i]; s += v; s2 += v * v; }
  rs[tid] = s; rs2[tid] = s2;
  __syncthreads();
  for (int off = 128; off > 0; off >>= 1) {
    if (tid < off) { rs[tid] += rs[tid + off]; rs2[tid] += rs2[tid + off]; }
    __syncthreads();
  }
  float mean = rs[0] * (1.0f / D_EMB);
  float var  = (rs2[0] - (float)D_EMB * mean * mean) * (1.0f / (D_EMB - 1));
  float rstd = rsqrtf(var + EPS);
  bf16* orow = out + ((size_t)row << 10);
  for (int i = tid; i < D_EMB; i += 256)
    orow[i] = (bf16)(gamma[i] * (xr[i] - mean) * rstd + beta[i]);
}

// ---------------- unified projection GEMM (Q, K, or V) ----------------
// M=32 per wave, N=DH=64, K=D_EMB. Ping-pong double buffering (unroll x2):
// each half-iteration consumes one operand set while loads refill the other,
// so the loop-carried set is defined directly by loads (no register copies).
// transposed=0: Out[bh][t][e] row-major; transposed=1: Out[bh][e][t] (for V).

__global__ __launch_bounds__(32) void proj_gemm(const bf16* __restrict__ hB,
                                                const bf16* __restrict__ Wt,
                                                bf16* __restrict__ Out,
                                                int transposed) {
  const int RT = TSEQ / 32;                  // 64 row tiles per (b,h)
  int blk = blockIdx.x;
  int rt = blk % RT;
  int bh = blk / RT;                         // 0..31
  int b = bh >> 4;
  int h = bh & 15;
  int lane = threadIdx.x, hi = lane >> 4, lo = lane & 15;

  const bf16* arow0 = hB + (((size_t)b * TSEQ + rt * 32 + lo) << 10);
  const bf16* arow1 = arow0 + ((size_t)16 << 10);
  const bf16* wbase = Wt + (((size_t)(h * DH + lo)) << 10) + hi * 16;

  v8f acc[2][4];
#pragma unroll
  for (int et = 0; et < 4; ++et) { acc[0][et] = vzero8(); acc[1][et] = vzero8(); }

  auto loadset = [&](int k) {
    OpSet s;
    s.a0 = load_a16(arow0 + k, hi);
    s.a1 = load_a16(arow1 + k, hi);
#pragma unroll
    for (int et = 0; et < 4; ++et)
      s.b[et] = *(const v16bf*)(wbase + (((size_t)et * 16) << 10) + k);
    return s;
  };
  auto domac = [&](const OpSet& s) {
#pragma unroll
    for (int et = 0; et < 4; ++et) {
      acc[0][et] = wmma_bf16(s.a0, s.b[et], acc[0][et]);
      acc[1][et] = wmma_bf16(s.a1, s.b[et], acc[1][et]);
    }
  };

  OpSet s0 = loadset(0);
  for (int k0 = 0; k0 < D_EMB; k0 += 64) {
    OpSet s1 = loadset(k0 + 32);             // always in-bounds (K % 64 == 0)
    domac(s0);
    s0 = loadset((k0 + 64) & (D_EMB - 1));   // wraps on final iter (discarded)
    domac(s1);
  }

  if (!transposed) {
#pragma unroll
    for (int half = 0; half < 2; ++half)
#pragma unroll
      for (int et = 0; et < 4; ++et) {
        int e = et * 16 + lo;
#pragma unroll
        for (int r = 0; r < 8; ++r) {
          size_t t = (size_t)rt * 32 + half * 16 + hi * 8 + r;
          Out[(((size_t)bh * TSEQ + t) << 6) + e] = (bf16)acc[half][et][r];
        }
      }
  } else {
#pragma unroll
    for (int half = 0; half < 2; ++half)
#pragma unroll
      for (int et = 0; et < 4; ++et) {
        int e = et * 16 + lo;
        v8bf vv;
#pragma unroll
        for (int r = 0; r < 8; ++r) vv[r] = (bf16)acc[half][et][r];
        *(v8bf*)(Out + (((size_t)bh * DH + e) << 11) +
                 (size_t)rt * 32 + half * 16 + hi * 8) = vv;
      }
  }
}

// ---------------- causal flash attention: one wave per 16-query tile --------

__global__ __launch_bounds__(32) void attn_fwd(const bf16* __restrict__ Q,
                                               const bf16* __restrict__ Kmat,
                                               const bf16* __restrict__ Vt,
                                               float* __restrict__ O) {
  __shared__ __align__(32) bf16 P[16][32];   // probabilities, C-layout -> A-layout bounce
  int blk = blockIdx.x;
  int tt = blk % (TSEQ / 16);
  int bh = blk / (TSEQ / 16);
  int lane = threadIdx.x, hi = lane >> 4, lo = lane & 15;
  int q0 = tt * 16;

  const bf16* qrow = Q + (((size_t)bh * TSEQ + q0 + lo) << 6);
  v16bf a0 = load_a16(qrow, hi);        // K-dim chunk d = 0..31
  v16bf a1 = load_a16(qrow + 32, hi);   // K-dim chunk d = 32..63

  float m[8], l[8];
  v8f accO[4];
#pragma unroll
  for (int r = 0; r < 8; ++r) { m[r] = -3.0e38f; l[r] = 0.0f; }
#pragma unroll
  for (int et = 0; et < 4; ++et) accO[et] = vzero8();

  int nkb = (q0 + 47) >> 5;            // ceil((q0+16)/32) 32-wide key blocks
  for (int kb = 0; kb < nkb; ++kb) {
    int kbase = kb * 32;

    // issue all K-operand loads, then the S WMMAs
    v16bf kb0[2], kb1[2];
#pragma unroll
    for (int j = 0; j < 2; ++j) {
      const bf16* krow = Kmat + (((size_t)bh * TSEQ + kbase + j * 16 + lo) << 6);
      kb0[j] = *(const v16bf*)(krow + hi * 16);
      kb1[j] = *(const v16bf*)(krow + 32 + hi * 16);
    }
    // V operands are independent of softmax math: issue now, consume later
    v16bf bv[4];
#pragma unroll
    for (int et = 0; et < 4; ++et) {
      const bf16* vr = Vt + (((size_t)bh * DH + et * 16 + lo) << 11) + kbase + hi * 16;
      bv[et] = *(const v16bf*)vr;
    }

    v8f S[2];
#pragma unroll
    for (int j = 0; j < 2; ++j) {
      v8f s = vzero8();
      s = wmma_bf16(a0, kb0[j], s);
      s = wmma_bf16(a1, kb1[j], s);
      S[j] = s;
    }

    // online softmax update (rows live at element r, row = q0 + 8*hi + r)
    float p0v[8], p1v[8];
#pragma unroll
    for (int r = 0; r < 8; ++r) {
      int row = q0 + hi * 8 + r;
      float s0 = S[0][r] * 0.03125f;             // 1/sqrt(1024)
      float s1 = S[1][r] * 0.03125f;
      if (kbase + lo > row)      s0 = -3.0e38f;  // causal mask
      if (kbase + 16 + lo > row) s1 = -3.0e38f;
      float mx = redmax16(fmaxf(s0, s1));
      float mn = fmaxf(m[r], mx);
      float alpha = __expf(m[r] - mn);
      float p0 = __expf(s0 - mn);
      float p1 = __expf(s1 - mn);
      l[r] = l[r] * alpha + redsum16(p0 + p1);
      m[r] = mn;
#pragma unroll
      for (int et = 0; et < 4; ++et) accO[et][r] *= alpha;
      p0v[r] = p0; p1v[r] = p1;
    }

    // C-layout -> LDS -> A-layout for the P (16x32) operand
#pragma unroll
    for (int r = 0; r < 8; ++r) {
      P[hi * 8 + r][lo]      = (bf16)p0v[r];
      P[hi * 8 + r][16 + lo] = (bf16)p1v[r];
    }
    asm volatile("s_wait_dscnt 0x0" ::: "memory");
    v8bf plo = *(const v8bf*)&P[lo][hi * 16];
    v8bf phi = *(const v8bf*)&P[lo][hi * 16 + 8];
    v16bf pa;
#pragma unroll
    for (int i = 0; i < 8; ++i) { pa[i] = plo[i]; pa[i + 8] = phi[i]; }

    // accO += P * V  (V stored transposed: column e contiguous over keys)
#pragma unroll
    for (int et = 0; et < 4; ++et) accO[et] = wmma_bf16(pa, bv[et], accO[et]);
  }

#pragma unroll
  for (int et = 0; et < 4; ++et) {
#pragma unroll
    for (int r = 0; r < 8; ++r) {
      float inv_l = 1.0f / l[r];
      O[(((size_t)bh * TSEQ + q0 + hi * 8 + r) << 6) + et * 16 + lo] = accO[et][r] * inv_l;
    }
  }
}

// ---------------- residual add + LayerNorm 2 ----------------

__global__ __launch_bounds__(256) void resid_ln2(const float* __restrict__ x,
                                                 const float* __restrict__ O,
                                                 const float* __restrict__ gamma,
                                                 const float* __restrict__ beta,
                                                 float* __restrict__ resid,
                                                 bf16* __restrict__ h2) {
  __shared__ float buf[D_EMB];
  __shared__ float rs[256], rs2[256];
  int row = blockIdx.x;                 // 0..BT-1
  int tid = threadIdx.x;
  int b = row >> 11;
  int t = row & 2047;
  float s = 0.f, s2 = 0.f;
  for (int i = tid; i < D_EMB; i += 256) {
    int h = i >> 6, e = i & 63;
    float v = x[((size_t)row << 10) + i] +
              O[((((size_t)(b * NH + h)) * TSEQ + t) << 6) + e];
    buf[i] = v; s += v; s2 += v * v;
  }
  rs[tid] = s; rs2[tid] = s2;
  __syncthreads();
  for (int off = 128; off > 0; off >>= 1) {
    if (tid < off) { rs[tid] += rs[tid + off]; rs2[tid] += rs2[tid + off]; }
    __syncthreads();
  }
  float mean = rs[0] * (1.0f / D_EMB);
  float var  = (rs2[0] - (float)D_EMB * mean * mean) * (1.0f / (D_EMB - 1));
  float rstd = rsqrtf(var + EPS);
  for (int i = tid; i < D_EMB; i += 256) {
    float v = buf[i];
    resid[((size_t)row << 10) + i] = v;
    h2[((size_t)row << 10) + i] = (bf16)(gamma[i] * (v - mean) * rstd + beta[i]);
  }
}

// ---------------- FFN GEMM 1: ff = relu(h2 @ W1 + b1) -> bf16 ----------------
// M=32 per wave, ping-pong double-buffered operands (no rotation copies).

__global__ __launch_bounds__(32) void ffn1_gemm(const bf16* __restrict__ h2,
                                                const bf16* __restrict__ W1t,
                                                const float* __restrict__ b1,
                                                bf16* __restrict__ ff) {
  int blk = blockIdx.x;
  int cg = blk % (DFF / 64);            // 64-wide column group
  int rt = blk / (DFF / 64);            // 32-row tile
  int lane = threadIdx.x, hi = lane >> 4, lo = lane & 15;

  const bf16* arow0 = h2 + (((size_t)rt * 32 + lo) << 10);
  const bf16* arow1 = arow0 + ((size_t)16 << 10);
  const bf16* wbase = W1t + (((size_t)(cg * 64 + lo)) << 10) + hi * 16;

  v8f acc[2][4];
#pragma unroll
  for (int et = 0; et < 4; ++et) { acc[0][et] = vzero8(); acc[1][et] = vzero8(); }

  auto loadset = [&](int k) {
    OpSet s;
    s.a0 = load_a16(arow0 + k, hi);
    s.a1 = load_a16(arow1 + k, hi);
#pragma unroll
    for (int et = 0; et < 4; ++et)
      s.b[et] = *(const v16bf*)(wbase + (((size_t)et * 16) << 10) + k);
    return s;
  };
  auto domac = [&](const OpSet& s) {
#pragma unroll
    for (int et = 0; et < 4; ++et) {
      acc[0][et] = wmma_bf16(s.a0, s.b[et], acc[0][et]);
      acc[1][et] = wmma_bf16(s.a1, s.b[et], acc[1][et]);
    }
  };

  OpSet s0 = loadset(0);
  for (int k0 = 0; k0 < D_EMB; k0 += 64) {
    OpSet s1 = loadset(k0 + 32);
    domac(s0);
    s0 = loadset((k0 + 64) & (D_EMB - 1));
    domac(s1);
  }

#pragma unroll
  for (int half = 0; half < 2; ++half) {
#pragma unroll
    for (int et = 0; et < 4; ++et) {
      int col = cg * 64 + et * 16 + lo;
      float bias = b1[col];
#pragma unroll
      for (int r = 0; r < 8; ++r) {
        float v = fmaxf(acc[half][et][r] + bias, 0.0f);
        ff[(((size_t)rt * 32 + half * 16 + hi * 8 + r) << 12) + col] = (bf16)v;
      }
    }
  }
}

// ---------------- FFN GEMM 2: out = resid + ff @ W2 + b2 (M=32) --------------

__global__ __launch_bounds__(32) void ffn2_gemm(const bf16* __restrict__ ff,
                                                const bf16* __restrict__ W2t,
                                                const float* __restrict__ b2,
                                                float* __restrict__ out) {
  int blk = blockIdx.x;
  int cg = blk % (D_EMB / 64);
  int rt = blk / (D_EMB / 64);          // 32-row tile
  int lane = threadIdx.x, hi = lane >> 4, lo = lane & 15;

  const bf16* arow0 = ff + (((size_t)rt * 32 + lo) << 12);
  const bf16* arow1 = arow0 + ((size_t)16 << 12);
  const bf16* wbase = W2t + (((size_t)(cg * 64 + lo)) << 12) + hi * 16;

  v8f acc[2][4];
#pragma unroll
  for (int et = 0; et < 4; ++et) { acc[0][et] = vzero8(); acc[1][et] = vzero8(); }

  auto loadset = [&](int k) {
    OpSet s;
    s.a0 = load_a16(arow0 + k, hi);
    s.a1 = load_a16(arow1 + k, hi);
#pragma unroll
    for (int et = 0; et < 4; ++et)
      s.b[et] = *(const v16bf*)(wbase + (((size_t)et * 16) << 12) + k);
    return s;
  };
  auto domac = [&](const OpSet& s) {
#pragma unroll
    for (int et = 0; et < 4; ++et) {
      acc[0][et] = wmma_bf16(s.a0, s.b[et], acc[0][et]);
      acc[1][et] = wmma_bf16(s.a1, s.b[et], acc[1][et]);
    }
  };

  OpSet s0 = loadset(0);
  for (int k0 = 0; k0 < DFF; k0 += 64) {
    OpSet s1 = loadset(k0 + 32);
    domac(s0);
    s0 = loadset((k0 + 64) & (DFF - 1));
    domac(s1);
  }

#pragma unroll
  for (int half = 0; half < 2; ++half) {
#pragma unroll
    for (int et = 0; et < 4; ++et) {
      int col = cg * 64 + et * 16 + lo;
      float bias = b2[col];
#pragma unroll
      for (int r = 0; r < 8; ++r) {
        size_t idx = (((size_t)rt * 32 + half * 16 + hi * 8 + r) << 10) + col;
        out[idx] = out[idx] + acc[half][et][r] + bias;   // resid already in out
      }
    }
  }
}

// ---------------- host launcher ----------------

extern "C" void kernel_launch(void* const* d_in, const int* in_sizes, int n_in,
                              void* d_out, int out_size, void* d_ws, size_t ws_size,
                              hipStream_t stream) {
  const float* x      = (const float*)d_in[0];
  const float* Wq     = (const float*)d_in[1];
  const float* Wk     = (const float*)d_in[2];
  const float* Wv     = (const float*)d_in[3];
  const float* W1     = (const float*)d_in[4];
  const float* b1     = (const float*)d_in[5];
  const float* W2     = (const float*)d_in[6];
  const float* b2     = (const float*)d_in[7];
  const float* gamma1 = (const float*)d_in[8];
  const float* beta1  = (const float*)d_in[9];
  const float* gamma2 = (const float*)d_in[10];
  const float* beta2  = (const float*)d_in[11];
  float* out = (float*)d_out;

  char* ws = (char*)d_ws;
  size_t off = 0;
  auto carve = [&](size_t bytes) -> void* {
    void* p = ws + off;
    off = (off + bytes + 255) & ~(size_t)255;
    return p;
  };

  bf16* hB   = (bf16*)carve((size_t)BT * D_EMB * 2);          // 8 MB
  bf16* WqT  = (bf16*)carve((size_t)NH * DH * D_EMB * 2);     // 2 MB
  bf16* WkT  = (bf16*)carve((size_t)NH * DH * D_EMB * 2);
  bf16* WvT  = (bf16*)carve((size_t)NH * DH * D_EMB * 2);
  bf16* Qb   = (bf16*)carve((size_t)NB * NH * TSEQ * DH * 2); // 8 MB
  bf16* Kb   = (bf16*)carve((size_t)NB * NH * TSEQ * DH * 2);
  bf16* VtB  = (bf16*)carve((size_t)NB * NH * DH * TSEQ * 2);
  float* Ob  = (float*)carve((size_t)NB * NH * TSEQ * DH * 4);// 16 MB
  bf16* h2B  = (bf16*)carve((size_t)BT * D_EMB * 2);
  bf16* W1t  = (bf16*)carve((size_t)DFF * D_EMB * 2);         // 8 MB
  bf16* W2t  = (bf16*)carve((size_t)D_EMB * DFF * 2);         // 8 MB
  bf16* ffB  = (bf16*)carve((size_t)BT * DFF * 2);            // 32 MB
  (void)ws_size; (void)in_sizes; (void)n_in; (void)out_size;

  // weight prep
  cvt_qkv_t<<<(NH * D_EMB * DH + 255) / 256, 256, 0, stream>>>(Wq, WqT);
  cvt_qkv_t<<<(NH * D_EMB * DH + 255) / 256, 256, 0, stream>>>(Wk, WkT);
  cvt_qkv_t<<<(NH * D_EMB * DH + 255) / 256, 256, 0, stream>>>(Wv, WvT);
  cvt_w1_t<<<(D_EMB * DFF + 255) / 256, 256, 0, stream>>>(W1, W1t);
  cvt_w2_t<<<(DFF * D_EMB + 255) / 256, 256, 0, stream>>>(W2, W2t);

  // LN1
  ln1_fwd<<<BT, 256, 0, stream>>>(x, gamma1, beta1, hB);

  // Q / K / V projections (WMMA, ping-pong double-buffered)
  const int projBlocks = NB * NH * (TSEQ / 32);   // 2048
  proj_gemm<<<projBlocks, 32, 0, stream>>>(hB, WqT, Qb,  0);
  proj_gemm<<<projBlocks, 32, 0, stream>>>(hB, WkT, Kb,  0);
  proj_gemm<<<projBlocks, 32, 0, stream>>>(hB, WvT, VtB, 1);

  // causal flash attention (WMMA)
  attn_fwd<<<NB * NH * (TSEQ / 16), 32, 0, stream>>>(Qb, Kb, VtB, Ob);

  // residual + LN2 (writes resid into d_out)
  resid_ln2<<<BT, 256, 0, stream>>>(x, Ob, gamma2, beta2, out, h2B);

  // FFN (WMMA, ping-pong double-buffered)
  ffn1_gemm<<<(BT / 32) * (DFF / 64), 32, 0, stream>>>(h2B, W1t, b1, ffB);
  ffn2_gemm<<<(BT / 32) * (D_EMB / 64), 32, 0, stream>>>(ffB, W2t, b2, out);
}
